// LinearAttention_64424509440481
// MI455X (gfx1250) — compile-verified, matrix operand order checked by reading, not run on hardware
//
#include <hip/hip_runtime.h>
#include <hip/hip_bf16.h>

// ---------------------------------------------------------------------------
// Linear attention (phi = elu+1) for B=4, S=4096, DIM=1024, H=16, Dh=64.
// Heavy math on v_wmma_f32_16x16x32_f16. Projection GEMMs stage tiles via
// the CDNA5 Tensor Data Mover (tensor_load_to_lds + s_wait_tensorcnt) with
// LDS ping-pong; qkv stages kv_ext via async global->LDS (ASYNCcnt).
// ---------------------------------------------------------------------------

typedef __attribute__((ext_vector_type(16))) _Float16 v16h;
typedef __attribute__((ext_vector_type(8)))  _Float16 v8h;
typedef __attribute__((ext_vector_type(4)))  _Float16 v4h;
typedef __attribute__((ext_vector_type(8)))  float    v8f;
typedef __attribute__((ext_vector_type(4)))  unsigned tdm_v4u;
typedef __attribute__((ext_vector_type(8)))  int      tdm_v8i;
typedef __attribute__((ext_vector_type(4)))  int      tdm_v4i;

#define DIM_ 1024
#define S_   4096
#define B_   4
#define H_   16
#define BS_  (B_ * H_ * 256)    // 16384 rows  (= 4*4096)
#define KVLD 64                 // kvT row length
#define KVROWS 80               // 64 + normalizer row + pad rows
#define TLD 40                  // padded LDS row length f16 (80B, 16B aligned)
#define QLD 72                  // qkv kernel LDS row stride (144B, 16B aligned)

#define HAVE_TDM __has_builtin(__builtin_amdgcn_tensor_load_to_lds)

// ---- WMMA wrapper ----------------------------------------------------------
__device__ __forceinline__ v8f wmma32(v16h a, v16h b, v8f c) {
  return __builtin_amdgcn_wmma_f32_16x16x32_f16(false, a, false, b, (short)0, c,
                                                false, false);
}

// ---- fragment load: row-major tile, lane-dim contiguous in K ---------------
//   lane<16: K = {0..7, 16..23}; lane>=16: K = {8..15, 24..31}
__device__ __forceinline__ v16h load_frag(const _Float16* tile, int ld) {
  int lane = threadIdx.x & 31;
  const _Float16* p = tile + (lane & 15) * ld + ((lane >> 4) << 3);
  v8h lo = *(const v8h*)(p);        // K base .. base+7   (16B)
  v8h hi = *(const v8h*)(p + 16);   // K base+16 .. +23   (16B)
  v16h f;
#pragma unroll
  for (int i = 0; i < 8; ++i) { f[i] = lo[i]; f[i + 8] = hi[i]; }
  return f;
}

// ---- CDNA5 async global->LDS 16B copy (per lane), ASYNCcnt-tracked ---------
__device__ __forceinline__ void async_copy16(const _Float16* gsrc,
                                             _Float16* ldst) {
  unsigned long long ga = (unsigned long long)(__UINTPTR_TYPE__)gsrc;
  unsigned la = (unsigned)(unsigned long long)(__UINTPTR_TYPE__)ldst;
  asm volatile("global_load_async_to_lds_b128 %0, %1, off"
               :: "v"(la), "v"(ga) : "memory");
}
__device__ __forceinline__ void wait_async0() {
  asm volatile("s_wait_asynccnt 0" ::: "memory");
}

// ---- CDNA5 TDM: 2D tile global->LDS with LDS row padding -------------------
// D# per ISA 8.3/8.4. data_size=2B; pad 4 DWORDs (16B) every 16 DWORDs (64B)
// of stored data -> LDS row stride 80B for a 32-elem (64B) tile row.
__device__ __forceinline__ void tdm_load_2d(const _Float16* gsrc,
                                            unsigned lds_off,
                                            unsigned tensor_d0,
                                            unsigned tensor_d1,
                                            unsigned tile_d0, unsigned tile_d1,
                                            unsigned stride0_elems) {
#if HAVE_TDM
  unsigned long long ga = (unsigned long long)(__UINTPTR_TYPE__)gsrc;
  tdm_v4u g0;
  g0[0] = 1u;                                   // count=1 (valid), user mode
  g0[1] = lds_off;                              // lds_addr (bytes)
  g0[2] = (unsigned)(ga & 0xFFFFFFFFull);       // global_addr[31:0]
  g0[3] = (unsigned)((ga >> 32) & 0x1FFFFFFull) // global_addr[56:32]
          | (2u << 30);                         // type = 2 ("image")
  tdm_v8i g1;
  g1[0] = (int)((1u << 16)      // data_size = 1 -> 2 bytes
              | (1u << 20)      // pad_enable
              | (3u << 22)      // pad_interval: 16 DWORDs of data
              | (3u << 25));    // pad_amount:   4 DWORDs of pad
  g1[1] = (int)((tensor_d0 & 0xFFFFu) << 16);              // dim0[15:0] @48
  g1[2] = (int)(((tensor_d0 >> 16) & 0xFFFFu)              // dim0[31:16]
              | ((tensor_d1 & 0xFFFFu) << 16));            // dim1[15:0]
  g1[3] = (int)(((tensor_d1 >> 16) & 0xFFFFu)              // dim1[31:16]
              | ((tile_d0 & 0xFFFFu) << 16));              // tile_dim0
  g1[4] = (int)(tile_d1 & 0xFFFFu);                        // tile_dim1; d2=0
  g1[5] = (int)stride0_elems;                              // dim0_stride[31:0]
  g1[6] = 0;                                               // stride0 hi, s1 lo
  g1[7] = 0;
  tdm_v4i z4 = {0, 0, 0, 0};
  tdm_v8i z8 = {0, 0, 0, 0, 0, 0, 0, 0};
  __builtin_amdgcn_tensor_load_to_lds(g0, g1, z4, z4, z8, 0);
#else
  (void)gsrc; (void)lds_off; (void)tensor_d0; (void)tensor_d1;
  (void)tile_d0; (void)tile_d1; (void)stride0_elems;
#endif
}

// ---------------------------------------------------------------------------
// Kernel 0: fp32 -> fp16 convert, 4-wide
// ---------------------------------------------------------------------------
__global__ void cvt_f32_f16_v4(const float* __restrict__ src,
                               _Float16* __restrict__ dst, int n4) {
  int i = blockIdx.x * blockDim.x + threadIdx.x;
  if (i < n4) {
    float4 f = ((const float4*)src)[i];
    v4h h;
    h[0] = (_Float16)f.x; h[1] = (_Float16)f.y;
    h[2] = (_Float16)f.z; h[3] = (_Float16)f.w;
    *(v4h*)(dst + 4 * (size_t)i) = h;
  }
}

// ---------------------------------------------------------------------------
// Kernel 1: projection GEMM  out = epilogue(A @ W^T + bias)
//   MODE 0: q = phi(.)      MODE 1: k = phi(.)*mask
//   MODE 2: v = (.)*mask    MODE 3: out = (.)  -> f32 (final projection)
// Block tile 128x128, 256 threads = 8 waves (2M x 4N), each wave 64x32.
// K-step 32; tiles staged by TDM (wave 0 issues two descriptors), LDS
// ping-pong, one s_wait_tensorcnt + barrier per step.
// ---------------------------------------------------------------------------
template <int MODE>
__global__ __launch_bounds__(256) void gemm_proj(
    const _Float16* __restrict__ A, const _Float16* __restrict__ W,
    const float* __restrict__ bias, const float* __restrict__ mask,
    _Float16* __restrict__ outH, float* __restrict__ outF) {
  __shared__ _Float16 sA[2][128 * TLD];   // 10,240 B each
  __shared__ _Float16 sB[2][128 * TLD];

  const int tid  = threadIdx.x;
  const int lane = tid & 31;
  const int wave = tid >> 5;
  const int waveM = wave >> 2;            // 0..1
  const int waveN = wave & 3;             // 0..3
  const int blockRow = blockIdx.y * 128;
  const int blockCol = blockIdx.x * 128;

  auto stage = [&](int buf, int k0) {
#if HAVE_TDM
    if (wave == 0) {
      tdm_load_2d(A + (size_t)blockRow * DIM_ + k0,
                  (unsigned)(unsigned long long)(__UINTPTR_TYPE__)&sA[buf][0],
                  DIM_, BS_, 32, 128, DIM_);
      tdm_load_2d(W + (size_t)blockCol * DIM_ + k0,
                  (unsigned)(unsigned long long)(__UINTPTR_TYPE__)&sB[buf][0],
                  DIM_, DIM_, 32, 128, DIM_);
    }
#else
#pragma unroll
    for (int i = 0; i < 2; ++i) {
      const int c   = tid + i * 256;      // chunk 0..511
      const int row = c >> 2;             // 0..127
      const int off = (c & 3) * 8;        // f16 elems: 0,8,16,24
      async_copy16(A + (size_t)(blockRow + row) * DIM_ + k0 + off,
                   &sA[buf][row * TLD + off]);
      async_copy16(W + (size_t)(blockCol + row) * DIM_ + k0 + off,
                   &sB[buf][row * TLD + off]);
    }
#endif
  };

  v8f acc[4][2] = {};
  stage(0, 0);
  for (int ks = 0; ks < DIM_ / 32; ++ks) {
#if HAVE_TDM
    if (wave == 0) __builtin_amdgcn_s_wait_tensorcnt(0);
#else
    wait_async0();
#endif
    __syncthreads();                      // tiles landed; prev reads done
    if (ks + 1 < DIM_ / 32) stage((ks + 1) & 1, (ks + 1) * 32);

    const _Float16* bufA = &sA[ks & 1][0];
    const _Float16* bufB = &sB[ks & 1][0];
    v16h a[4], b[2];
#pragma unroll
    for (int mt = 0; mt < 4; ++mt)
      a[mt] = load_frag(bufA + (waveM * 64 + mt * 16) * TLD, TLD);
#pragma unroll
    for (int nt = 0; nt < 2; ++nt)
      b[nt] = load_frag(bufB + (waveN * 32 + nt * 16) * TLD, TLD);
#pragma unroll
    for (int mt = 0; mt < 4; ++mt)
#pragma unroll
      for (int nt = 0; nt < 2; ++nt)
        acc[mt][nt] = wmma32(a[mt], b[nt], acc[mt][nt]);
  }

  const int n  = lane & 15;
  const int mb = (lane >> 4) << 3;        // 0 or 8
#pragma unroll
  for (int mt = 0; mt < 4; ++mt) {
#pragma unroll
    for (int nt = 0; nt < 2; ++nt) {
      const int col = blockCol + waveN * 32 + nt * 16 + n;
      const float bv = bias[col];
#pragma unroll
      for (int r = 0; r < 8; ++r) {
        const int row = blockRow + waveM * 64 + mt * 16 + mb + r;
        float v = acc[mt][nt][r] + bv;
        if (MODE == 0 || MODE == 1) v = (v > 0.f) ? (v + 1.f) : __expf(v);
        if (MODE == 1 || MODE == 2) v *= mask[row];
        if (MODE == 3)
          outF[(size_t)row * DIM_ + col] = v;
        else
          outH[(size_t)row * DIM_ + col] = (_Float16)v;
      }
    }
  }
}

// ---------------------------------------------------------------------------
// Kernel 2: kv_ext per (b,h):  kvT[bh][m'][d], m' in 0..79:
//   m' < 64 : kv^T = sum_s v[s,m'] * k[s,d]
//   m' == 64: k_sum (ones-row A fragment, extra accumulator on rt==0 waves)
//   m' > 64 : zeros
// One workgroup per (b,h); 512 threads = 16 waves = 4x4 tiles of 64x64,
// straight-line transposed staging of 32-s chunks into padded LDS.
// ---------------------------------------------------------------------------
__global__ __launch_bounds__(512) void kv_kernel(
    const _Float16* __restrict__ kh, const _Float16* __restrict__ vh,
    _Float16* __restrict__ kvT) {
  const int bh = blockIdx.x;
  const int b  = bh >> 4;
  const int h  = bh & 15;
  __shared__ _Float16 kT[64][TLD];
  __shared__ _Float16 vT[64][TLD];

  const int tid  = threadIdx.x;
  const int lane = tid & 31;
  const int wave = tid >> 5;          // 0..15
  const int ct   = wave & 3;          // d tile
  const int rt   = wave >> 2;         // m' tile 0..3

  const int ss = tid >> 4;            // 0..31 (s within chunk)
  const int dd = (tid & 15) * 4;      // 0..60
  const size_t headOff = (size_t)b * S_ * DIM_ + (size_t)h * 64;

  v16h aOnes;                         // row 0 = 1.0 -> k_sum row
  {
    const _Float16 e = ((lane & 15) == 0) ? (_Float16)1.0f : (_Float16)0.0f;
#pragma unroll
    for (int i = 0; i < 16; ++i) aOnes[i] = e;
  }

  v8f acc = {};
  v8f accS = {};
  for (int s0 = 0; s0 < S_; s0 += 32) {
    __syncthreads();                  // previous chunk fully consumed
    {
      const size_t rowOff = headOff + (size_t)(s0 + ss) * DIM_ + dd;
      v4h xk = *(const v4h*)(kh + rowOff);
      v4h xv = *(const v4h*)(vh + rowOff);
#pragma unroll
      for (int j = 0; j < 4; ++j) {
        kT[dd + j][ss] = xk[j];
        vT[dd + j][ss] = xv[j];
      }
    }
    __syncthreads();

    v16h a  = load_frag(&vT[rt * 16][0], TLD);
    v16h bf = load_frag(&kT[ct * 16][0], TLD);
    acc = wmma32(a, bf, acc);
    if (rt == 0)                      // wave-uniform branch
      accS = wmma32(aOnes, bf, accS);
  }

  const int n  = lane & 15;
  const int mb = (lane >> 4) << 3;
  _Float16* out = kvT + (size_t)bh * KVROWS * KVLD;
#pragma unroll
  for (int r = 0; r < 8; ++r)
    out[(rt * 16 + mb + r) * KVLD + ct * 16 + n] = (_Float16)acc[r];
  if (rt == 0) {
#pragma unroll
    for (int r = 0; r < 8; ++r)
      out[(64 + mb + r) * KVLD + ct * 16 + n] = (_Float16)accS[r];
  }
}

// ---------------------------------------------------------------------------
// Kernel 3: qkv_ext = q @ kv_ext, normalize, -> attn (f16).
//   Per (b,h): M=4096, K=64, N=80; kv_ext staged once per block into LDS
//   via async global->LDS (conflict-free 144B row stride). N-column 64 is
//   the normalizer, broadcast within the wave via __shfl on acc[4].
// ---------------------------------------------------------------------------
__global__ __launch_bounds__(256) void qkv_kernel(
    const _Float16* __restrict__ qh, const _Float16* __restrict__ kvT,
    _Float16* __restrict__ attn) {
  __shared__ _Float16 sKV[KVROWS * QLD];   // 11,520 B
  const int bh   = blockIdx.y;
  const int h    = bh & 15;
  const int tid  = threadIdx.x;
  const int lane = tid & 31;
  const int wave = tid >> 5;
  const int row0 = blockIdx.x * 128 + wave * 16;           // s
  const size_t rowG = (size_t)(bh >> 4) * S_ + row0;       // b*S + s

  const _Float16* Abase = qh + rowG * DIM_ + (size_t)h * 64;
  const _Float16* Bbase = kvT + (size_t)bh * KVROWS * KVLD;

  // stage 80x64 kv_ext into LDS: 640 16B chunks
  for (int c = tid; c < KVROWS * 8; c += 256) {
    const int row = c >> 3;
    const int seg = c & 7;
    async_copy16(Bbase + row * KVLD + seg * 8, &sKV[row * QLD + seg * 8]);
  }
  wait_async0();
  __syncthreads();

  v8f acc[5] = {};
#pragma unroll
  for (int k0 = 0; k0 < 64; k0 += 32) {
    v16h a = load_frag(Abase + k0, DIM_);
#pragma unroll
    for (int nt = 0; nt < 5; ++nt) {
      v16h b = load_frag(&sKV[(nt * 16) * QLD + k0], QLD);
      acc[nt] = wmma32(a, b, acc[nt]);
    }
  }

  const int n  = lane & 15;
  const int mb = (lane >> 4) << 3;
#pragma unroll
  for (int r = 0; r < 8; ++r) {
    const float nrm = __shfl(acc[4][r], (lane & 16), 32);
    const float rcp = 1.0f / (nrm + 1e-6f);
    const size_t orow = (rowG + mb + r) * DIM_ + (size_t)h * 64;
#pragma unroll
    for (int nt = 0; nt < 4; ++nt)
      attn[orow + nt * 16 + n] = (_Float16)(acc[nt][r] * rcp);
  }
}

// ---------------------------------------------------------------------------
// launch
// ---------------------------------------------------------------------------
extern "C" void kernel_launch(void* const* d_in, const int* in_sizes, int n_in,
                              void* d_out, int out_size, void* d_ws,
                              size_t ws_size, hipStream_t stream) {
  const float* x    = (const float*)d_in[0];
  const float* mask = (const float*)d_in[1];
  const float* Wq   = (const float*)d_in[2];
  const float* bq   = (const float*)d_in[3];
  const float* Wk   = (const float*)d_in[4];
  const float* bk   = (const float*)d_in[5];
  const float* Wv   = (const float*)d_in[6];
  const float* bv   = (const float*)d_in[7];
  const float* Wo   = (const float*)d_in[8];
  const float* bo   = (const float*)d_in[9];
  float* out = (float*)d_out;

  _Float16* ws = (_Float16*)d_ws;
  const size_t SZ  = (size_t)BS_ * DIM_;       // 16,777,216 f16 elems
  const size_t WSZ = (size_t)DIM_ * DIM_;      // 1,048,576
  _Float16* xh  = ws;                          // reused as attn buffer later
  _Float16* qh  = ws + SZ;
  _Float16* kh  = ws + 2 * SZ;
  _Float16* vh  = ws + 3 * SZ;
  _Float16* wqh = ws + 4 * SZ;
  _Float16* wkh = wqh + WSZ;
  _Float16* wvh = wkh + WSZ;
  _Float16* woh = wvh + WSZ;
  _Float16* kvT = woh + WSZ;                   // B_*H_*80*64 f16

  // 0) converts (4-wide)
  cvt_f32_f16_v4<<<(int)(SZ / 4 / 256), 256, 0, stream>>>(x, xh, (int)(SZ / 4));
  cvt_f32_f16_v4<<<(int)(WSZ / 4 / 256), 256, 0, stream>>>(Wq, wqh, (int)(WSZ / 4));
  cvt_f32_f16_v4<<<(int)(WSZ / 4 / 256), 256, 0, stream>>>(Wk, wkh, (int)(WSZ / 4));
  cvt_f32_f16_v4<<<(int)(WSZ / 4 / 256), 256, 0, stream>>>(Wv, wvh, (int)(WSZ / 4));
  cvt_f32_f16_v4<<<(int)(WSZ / 4 / 256), 256, 0, stream>>>(Wo, woh, (int)(WSZ / 4));

  // 1) fused projections (phi/mask epilogues)
  dim3 ggrid(DIM_ / 128, BS_ / 128);           // (8, 128)
  gemm_proj<0><<<ggrid, 256, 0, stream>>>(xh, wqh, bq, mask, qh, nullptr);
  gemm_proj<1><<<ggrid, 256, 0, stream>>>(xh, wkh, bk, mask, kh, nullptr);
  gemm_proj<2><<<ggrid, 256, 0, stream>>>(xh, wvh, bv, mask, vh, nullptr);

  // 2) kv (+k_sum) per head
  kv_kernel<<<B_ * H_, 512, 0, stream>>>(kh, vh, kvT);

  // 3) qkv + normalization -> attn (reuses xh; x no longer needed)
  qkv_kernel<<<dim3(S_ / 128, B_ * H_), 256, 0, stream>>>(qh, kvT, xh);

  // 4) output projection, fp32 to d_out
  gemm_proj<3><<<ggrid, 256, 0, stream>>>(xh, woh, bo, mask, nullptr, out);
}
